// QuantPCAMatMul_14207751815304
// MI455X (gfx1250) — matrix operand order repaired and checked by reading, not compiled, not run on hardware
//
#include <hip/hip_runtime.h>
#include <hip/hip_fp16.h>

// ---- CDNA5 / gfx1250: wave32, WMMA f32_16x16x32_f16 ----
typedef __attribute__((ext_vector_type(16))) _Float16 v16h;
typedef __attribute__((ext_vector_type(8)))  _Float16 v8h;
typedef __attribute__((ext_vector_type(8)))  float    v8f;

#define HEADS    32
#define HEAD_DIM 128
#define B_DIM    4
#define T_DIM    1024
#define BH       (B_DIM * HEADS)        // 128 (b,h) slabs
#define SLAB     (T_DIM * HEAD_DIM)     // 131072 elems per slab (2^17)
#define LDS_PITCH 136                   // halves; 272B row stride -> conflict-free frag reads

// ------------------------------------------------------------------
// per-(b,h) amax of |x1|  (one block per slab, direct write)
// ------------------------------------------------------------------
__global__ void k_amax_x1(const float* __restrict__ x1, float* __restrict__ amax1) {
    __shared__ float red[256];
    const int bh = blockIdx.x;
    const float* p = x1 + (size_t)bh * SLAB;
    float m = 0.f;
    for (int i = threadIdx.x; i < SLAB; i += 256)
        m = fmaxf(m, fabsf(p[i]));
    red[threadIdx.x] = m;
    __syncthreads();
    for (int s = 128; s > 0; s >>= 1) {
        if ((int)threadIdx.x < s)
            red[threadIdx.x] = fmaxf(red[threadIdx.x], red[threadIdx.x + s]);
        __syncthreads();
    }
    if (threadIdx.x == 0) amax1[bh] = red[0];
}

__global__ void k_zero(float* __restrict__ p, int n) {
    int i = blockIdx.x * blockDim.x + threadIdx.x;
    if (i < n) p[i] = 0.f;
}

// ------------------------------------------------------------------
// x1 -> 8-bit qdq -> fp16
// ------------------------------------------------------------------
__global__ void k_qdq_x1(const float* __restrict__ x1,
                         const float* __restrict__ amax1,
                         _Float16* __restrict__ x1q) {
    const size_t n = (size_t)BH * SLAB;
    size_t stride = (size_t)gridDim.x * blockDim.x;
    for (size_t i = (size_t)blockIdx.x * blockDim.x + threadIdx.x; i < n; i += stride) {
        int bh = (int)(i >> 17);                       // / SLAB
        float scale = fmaxf(amax1[bh], 1e-8f) * (1.0f / 127.0f);
        float q = rintf(x1[i] / scale);                // round-to-nearest-even == jnp.round
        q = fminf(fmaxf(q, -128.f), 127.f);
        x1q[i] = (_Float16)(q * scale);
    }
}

// ------------------------------------------------------------------
// 128-pt orthonormal FWHT, wave32-native: 4 elems/lane (idx = lane + 32*j),
// distances 1..16 via __shfl_xor, distances 32/64 in registers.
// ------------------------------------------------------------------
__device__ inline void fwht128(float v[4], int lane) {
    #pragma unroll
    for (int d = 1; d <= 16; d <<= 1) {
        const bool upper = (lane & d) != 0;
        #pragma unroll
        for (int j = 0; j < 4; ++j) {
            float p = __shfl_xor(v[j], d, 32);
            v[j] = upper ? (p - v[j]) : (v[j] + p);
        }
    }
    float t0 = v[0], t1 = v[1], t2 = v[2], t3 = v[3];
    v[0] = t0 + t1; v[1] = t0 - t1; v[2] = t2 + t3; v[3] = t2 - t3;
    t0 = v[0]; t1 = v[1]; t2 = v[2]; t3 = v[3];
    v[0] = t0 + t2; v[2] = t0 - t2; v[1] = t1 + t3; v[3] = t1 - t3;
    const float s = 0.08838834764831845f;              // 1/sqrt(128)
    v[0] *= s; v[1] *= s; v[2] *= s; v[3] *= s;
}

// x2h = (x2 - mu) @ H ; track per-(b,h) amax (atomic max, non-negative bits)
__global__ void k_had_fwd(const float* __restrict__ x2,
                          const float* __restrict__ mu,
                          _Float16* __restrict__ x2h,
                          float* __restrict__ amax2) {
    const int bh   = blockIdx.x;
    const int h    = bh & (HEADS - 1);
    const int lane = threadIdx.x & 31;
    const int wave = threadIdx.x >> 5;
    const float* mup = mu + h * HEAD_DIM;
    const float m0 = mup[lane], m1 = mup[lane + 32], m2 = mup[lane + 64], m3 = mup[lane + 96];
    float wmax = 0.f;
    for (int i = 0; i < 16; ++i) {
        const int t = blockIdx.y * 128 + wave * 16 + i;
        const float* row = x2 + ((size_t)bh * T_DIM + t) * HEAD_DIM;
        float v[4] = { row[lane] - m0, row[lane + 32] - m1,
                       row[lane + 64] - m2, row[lane + 96] - m3 };
        fwht128(v, lane);
        _Float16* orow = x2h + ((size_t)bh * T_DIM + t) * HEAD_DIM;
        orow[lane]      = (_Float16)v[0];
        orow[lane + 32] = (_Float16)v[1];
        orow[lane + 64] = (_Float16)v[2];
        orow[lane + 96] = (_Float16)v[3];
        wmax = fmaxf(wmax, fmaxf(fmaxf(fabsf(v[0]), fabsf(v[1])),
                                 fmaxf(fabsf(v[2]), fabsf(v[3]))));
    }
    #pragma unroll
    for (int d = 16; d > 0; d >>= 1)
        wmax = fmaxf(wmax, __shfl_xor(wmax, d, 32));
    if (lane == 0)  // wmax >= 0: uint bit-order == float order
        atomicMax((unsigned int*)&amax2[bh], __float_as_uint(wmax));
}

// 4-bit qdq -> inverse Hadamard -> +mu, fp16 in place
__global__ void k_had_inv(_Float16* __restrict__ x2hq,
                          const float* __restrict__ mu,
                          const float* __restrict__ amax2) {
    const int bh   = blockIdx.x;
    const int h    = bh & (HEADS - 1);
    const int lane = threadIdx.x & 31;
    const int wave = threadIdx.x >> 5;
    const float scale = fmaxf(amax2[bh], 1e-8f) * (1.0f / 7.0f);
    const float inv   = 1.0f / scale;
    const float* mup = mu + h * HEAD_DIM;
    const float m0 = mup[lane], m1 = mup[lane + 32], m2 = mup[lane + 64], m3 = mup[lane + 96];
    for (int i = 0; i < 16; ++i) {
        const int t = blockIdx.y * 128 + wave * 16 + i;
        _Float16* row = x2hq + ((size_t)bh * T_DIM + t) * HEAD_DIM;
        float v[4] = { (float)row[lane], (float)row[lane + 32],
                       (float)row[lane + 64], (float)row[lane + 96] };
        #pragma unroll
        for (int j = 0; j < 4; ++j) {
            float q = rintf(v[j] * inv);
            q = fminf(fmaxf(q, -8.f), 7.f);
            v[j] = q * scale;
        }
        fwht128(v, lane);                               // H symmetric, H*H = I
        row[lane]      = (_Float16)(v[0] + m0);
        row[lane + 32] = (_Float16)(v[1] + m1);
        row[lane + 64] = (_Float16)(v[2] + m2);
        row[lane + 96] = (_Float16)(v[3] + m3);
    }
}

// ------------------------------------------------------------------
// Batched GEMM: out[bh] = X1q (1024x128) @ X2q^T (128x1024), f32 out.
// Block = 256 thr = 8 waves; block tile 128x128 (full K=128).
// A/B tiles staged once in LDS (coalesced b128 chunks), then fragments
// read via ds_load_b128 -> B tile fetched from global 1x per block
// instead of 8x (once per wave). Operands (64 MB) are L2-resident;
// output streamed with non-temporal stores.
// Fragment layouts per ISA 7.12.2:
//   A 16x32: lane l -> row (l&15); halves [ks+b .. +8) and [ks+b+16 .. +8), b = l<16?0:8
//   B 32x16: lane l -> col (l&15); halves [ks+b .. +8) and [ks+b+16 .. +8), same b as A
// ------------------------------------------------------------------
__global__ void __launch_bounds__(256, 2)
k_gemm(const _Float16* __restrict__ x1q,
       const _Float16* __restrict__ x2q,
       float* __restrict__ out) {
    __shared__ _Float16 Ash[128 * LDS_PITCH];   // 34 KB
    __shared__ _Float16 Bsh[128 * LDS_PITCH];   // 34 KB

    const int bh   = blockIdx.x;
    const int tm   = blockIdx.y * 128;
    const int tn   = blockIdx.z * 128;
    const int tid  = threadIdx.x;
    const int lane = tid & 31;
    const int wave = tid >> 5;
    const int lrow = lane & 15;
    const bool hi  = lane >= 16;

    // ---- cooperative global -> LDS stage (A and B tiles, 32 KB each) ----
    // 2048 16-byte chunks per tile; 8 chunks per thread, fully coalesced.
    const _Float16* Ag = x1q + ((size_t)bh * T_DIM + tm) * HEAD_DIM;
    const _Float16* Bg = x2q + ((size_t)bh * T_DIM + tn) * HEAD_DIM;
    #pragma unroll
    for (int i = 0; i < 8; ++i) {
        const int chunk = tid + 256 * i;          // 0..2047
        const int row   = chunk >> 4;             // 0..127
        const int c     = (chunk & 15) * 8;       // half offset within row
        *(v8h*)&Ash[row * LDS_PITCH + c] = *(const v8h*)(Ag + (size_t)row * HEAD_DIM + c);
        *(v8h*)&Bsh[row * LDS_PITCH + c] = *(const v8h*)(Bg + (size_t)row * HEAD_DIM + c);
    }
    __syncthreads();

    // ---- preload all 4 A fragments (rows wave*16 .. +15) into registers ----
    v16h Afrag[4];
    #pragma unroll
    for (int ksi = 0; ksi < 4; ++ksi) {
        const int ka = ksi * 32 + (hi ? 8 : 0);
        const _Float16* ap = &Ash[(wave * 16 + lrow) * LDS_PITCH + ka];
        v8h a0 = *(const v8h*)ap;                 // K = ka .. ka+7
        v8h a1 = *(const v8h*)(ap + 16);          // K = ka+16 .. ka+23
        Afrag[ksi] = __builtin_shufflevector(a0, a1, 0, 1, 2, 3, 4, 5, 6, 7,
                                                     8, 9, 10, 11, 12, 13, 14, 15);
    }

    // ---- 32 WMMAs: 8 N-tiles x 4 K-steps, B fragments from LDS ----
    v8f acc[8] = {};
    #pragma unroll
    for (int nt = 0; nt < 8; ++nt) {
        #pragma unroll
        for (int ksi = 0; ksi < 4; ++ksi) {
            const int kb = ksi * 32 + (hi ? 8 : 0);
            const _Float16* bp = &Bsh[(nt * 16 + lrow) * LDS_PITCH + kb];
            v8h b0 = *(const v8h*)bp;
            v8h b1 = *(const v8h*)(bp + 16);      // K = kb+16 .. kb+23
            v16h Bv = __builtin_shufflevector(b0, b1, 0, 1, 2, 3, 4, 5, 6, 7,
                                                      8, 9, 10, 11, 12, 13, 14, 15);
            acc[nt] = __builtin_amdgcn_wmma_f32_16x16x32_f16(
                false, Afrag[ksi], false, Bv, (short)0, acc[nt], false, false);
        }
    }

    // ---- store: lane l -> col (l&15); VGPR r -> row (l<16 ? r : 8+r) ----
    const int rbase = tm + wave * 16 + (hi ? 8 : 0);
    #pragma unroll
    for (int nt = 0; nt < 8; ++nt) {
        const int col = tn + nt * 16 + lrow;
        float* op = out + ((size_t)bh * T_DIM + rbase) * T_DIM + col;
        #pragma unroll
        for (int r = 0; r < 8; ++r)
            __builtin_nontemporal_store(acc[nt][r], op + (size_t)r * T_DIM);
    }
}

// ------------------------------------------------------------------
// Launch. Workspace layout (needs 64 MB + 1 KB):
//   [0, 32MB)        x1q  fp16
//   [32MB, 64MB)     x2h / x2q fp16 (transformed in place)
//   [64MB, +512B)    amax1 float[128]
//   [+512B, +1KB)    amax2 float[128]
// ------------------------------------------------------------------
extern "C" void kernel_launch(void* const* d_in, const int* in_sizes, int n_in,
                              void* d_out, int out_size, void* d_ws, size_t ws_size,
                              hipStream_t stream) {
    const float* x1 = (const float*)d_in[0];
    const float* x2 = (const float*)d_in[1];
    const float* mu = (const float*)d_in[2];
    float* out = (float*)d_out;

    char* ws = (char*)d_ws;
    _Float16* x1q  = (_Float16*)ws;
    _Float16* x2q  = (_Float16*)(ws + ((size_t)32 << 20));
    float*    amax1 = (float*)(ws + ((size_t)64 << 20));
    float*    amax2 = amax1 + 128;

    k_zero   <<<1, 256, 0, stream>>>(amax2, 128);      // ws poisoned 0xAA -> must zero for atomicMax
    k_amax_x1<<<BH, 256, 0, stream>>>(x1, amax1);
    k_qdq_x1 <<<2048, 256, 0, stream>>>(x1, amax1, x1q);
    k_had_fwd<<<dim3(BH, 8), 256, 0, stream>>>(x2, mu, x2q, amax2);
    k_had_inv<<<dim3(BH, 8), 256, 0, stream>>>(x2q, mu, amax2);
    k_gemm   <<<dim3(BH, 8, 8), 256, 0, stream>>>(x1q, x2q, out);
}
